// DeconvLayer_45586782880079
// MI455X (gfx1250) — matrix operand order repaired and checked by reading, not compiled, not run on hardware
//
#include <hip/hip_runtime.h>
#include <math.h>

// ---------------------------------------------------------------------------
// Problem constants (from reference): B=8, Cin=256, Cout=128, H=W=64
// ---------------------------------------------------------------------------
#define NB    8
#define CIN   256
#define COUT  128
#define HH    64
#define WW    64
#define P1    (HH*WW)          // 4096
#define KDIM  (CIN*9)          // 2304
#define HO    128
#define WO    128
#define P2    (HO*WO)          // 16384
#define EPS   1e-5f

typedef __bf16 v16bf __attribute__((ext_vector_type(16)));
typedef float  v8f   __attribute__((ext_vector_type(8)));

struct alignas(16) U4 { unsigned x, y, z, w; };
union AFrag { v16bf v; U4 q[2]; };
union H8    { U4 q; __bf16 e[8]; };

__device__ __forceinline__ __bf16 f2bf(float f) { return (__bf16)f; }

// ---------------------------------------------------------------------------
// 1) 3x3 offset conv: x[b,256,64,64] * off_w[27,256,3,3] + off_b -> om[b,27,64,64]
// ---------------------------------------------------------------------------
__global__ __launch_bounds__(256)
void offconv_kernel(const float* __restrict__ x, const float* __restrict__ off_w,
                    const float* __restrict__ off_b, float* __restrict__ om) {
  __shared__ float xs[18 * 18];
  const int b  = blockIdx.y;
  const int tileY = blockIdx.x >> 2, tileX = blockIdx.x & 3;
  const int tid = threadIdx.x;
  const int ty = tid >> 4, tx = tid & 15;
  const int h = tileY * 16 + ty, w = tileX * 16 + tx;

  float acc[27];
#pragma unroll
  for (int i = 0; i < 27; ++i) acc[i] = 0.f;

  for (int c = 0; c < CIN; ++c) {
    for (int i = tid; i < 324; i += 256) {
      int yy = i / 18, xx = i % 18;
      int gy = tileY * 16 + yy - 1, gx = tileX * 16 + xx - 1;
      float v = 0.f;
      if (gy >= 0 && gy < HH && gx >= 0 && gx < WW)
        v = x[(((size_t)b * CIN + c) * HH + gy) * WW + gx];
      xs[i] = v;
    }
    __syncthreads();
    float v[9];
#pragma unroll
    for (int k = 0; k < 9; ++k)
      v[k] = xs[(ty + k / 3) * 18 + tx + (k % 3)];
#pragma unroll
    for (int oc = 0; oc < 27; ++oc) {
      const float* wp = off_w + ((size_t)oc * CIN + c) * 9;  // uniform -> scalar loads
#pragma unroll
      for (int k = 0; k < 9; ++k) acc[oc] += wp[k] * v[k];
    }
    __syncthreads();
  }
#pragma unroll
  for (int oc = 0; oc < 27; ++oc)
    om[(((size_t)b * 27 + oc) * P1) + h * WW + w] = acc[oc] + off_b[oc];
}

// ---------------------------------------------------------------------------
// 2a) dcn_w f32[128][2304] -> bf16
// ---------------------------------------------------------------------------
__global__ void convert_w_kernel(const float* __restrict__ src, __bf16* __restrict__ dst, int n) {
  int i = blockIdx.x * 256 + threadIdx.x;
  if (i < n) dst[i] = f2bf(src[i]);
}

// ---------------------------------------------------------------------------
// 2b) deconv weights: 4 parity classes, w2[cls][o][K=c*4+t] bf16
// ---------------------------------------------------------------------------
__global__ void build_w2_kernel(const float* __restrict__ up_w, __bf16* __restrict__ w2) {
  int idx = blockIdx.x * 256 + threadIdx.x;      // 4*128*512 = 262144
  if (idx >= 4 * COUT * 512) return;
  int cls = idx >> 16;
  int rem = idx & 65535;
  int o = rem >> 9, K = rem & 511;
  int c = K >> 2, t = K & 3, tyi = t >> 1, txi = t & 1;
  int py = cls >> 1, px = cls & 1;
  int ky = ((py + 1) & 1) + 2 * tyi;
  int kx = ((px + 1) & 1) + 2 * txi;
  w2[idx] = f2bf(up_w[(((size_t)c * COUT + o) * 4 + ky) * 4 + kx]);
}

// ---------------------------------------------------------------------------
// 3) Deformable im2col for one batch: val[K=c*9+k2][p] bf16
// ---------------------------------------------------------------------------
__global__ __launch_bounds__(256)
void build_val_kernel(const float* __restrict__ x, const float* __restrict__ om,
                      __bf16* __restrict__ val, int b) {
  int gid = blockIdx.x * 256 + threadIdx.x;      // 9*4096 = 36864
  if (gid >= 9 * P1) return;
  int k2 = gid / P1, p = gid % P1;
  int h = p >> 6, w = p & 63;

  const float* omb = om + (size_t)b * 27 * P1;
  float dy  = omb[(2 * k2) * P1 + p];
  float dx  = omb[(2 * k2 + 1) * P1 + p];
  float mk  = omb[(18 + k2) * P1 + p];
  float msk = 1.f / (1.f + expf(-mk));

  float py = dy + (float)(k2 / 3 - 1) + (float)h;
  float px = dx + (float)(k2 % 3 - 1) + (float)w;
  float y0f = floorf(py), x0f = floorf(px);
  float ly = py - y0f, lx = px - x0f;
  int y0 = (int)y0f, x0 = (int)x0f;
  int y1 = y0 + 1,   x1 = x0 + 1;

  float v00 = (y0 >= 0 && y0 < HH && x0 >= 0 && x0 < WW) ? 1.f : 0.f;
  float v01 = (y0 >= 0 && y0 < HH && x1 >= 0 && x1 < WW) ? 1.f : 0.f;
  float v10 = (y1 >= 0 && y1 < HH && x0 >= 0 && x0 < WW) ? 1.f : 0.f;
  float v11 = (y1 >= 0 && y1 < HH && x1 >= 0 && x1 < WW) ? 1.f : 0.f;
  float w00 = (1.f - ly) * (1.f - lx) * v00 * msk;
  float w01 = (1.f - ly) * lx * v01 * msk;
  float w10 = ly * (1.f - lx) * v10 * msk;
  float w11 = ly * lx * v11 * msk;

  int cy0 = min(max(y0, 0), HH - 1), cy1 = min(max(y1, 0), HH - 1);
  int cx0 = min(max(x0, 0), WW - 1), cx1 = min(max(x1, 0), WW - 1);
  int i00 = cy0 * WW + cx0, i01 = cy0 * WW + cx1;
  int i10 = cy1 * WW + cx0, i11 = cy1 * WW + cx1;

  const float* xb = x + (size_t)b * CIN * P1;
  for (int c = 0; c < CIN; ++c) {
    const float* xc = xb + (size_t)c * P1;
    float v = w00 * xc[i00] + w01 * xc[i01] + w10 * xc[i10] + w11 * xc[i11];
    val[((size_t)(c * 9 + k2)) * P1 + p] = f2bf(v);
  }
}

// ---------------------------------------------------------------------------
// 4) DCN GEMM (one batch): y[o][p] = sum_K wbf[o][K] * val[K][p] + dcn_b[o]
//    Block tile 128x128, wave = 2 M-tiles x 4 N-tiles (8 accumulators),
//    B fragments reused across both A fragments. K-step 32 (72 steps).
// ---------------------------------------------------------------------------
__global__ __launch_bounds__(256)
void gemm_dcn_kernel(const __bf16* __restrict__ wbf, const __bf16* __restrict__ val,
                     const float* __restrict__ dcn_b, float* __restrict__ y, int b) {
  __shared__ __bf16 As[COUT * 32];    // 128 rows x 32 K      (8 KB)
  __shared__ __bf16 VsT[128 * 32];    // 128 cols x 32 K, K-contig (8 KB)

  const int tid  = threadIdx.x;
  const int lane = tid & 31;
  const int wv   = tid >> 5;
  const int wm   = wv & 3;                         // M pair: tiles 2*wm, 2*wm+1
  const int wn   = wv >> 2;                        // N half: tiles wn*4 .. wn*4+3
  const int nBase = blockIdx.x * 128;
  const int mRow0 = wm * 32 + (lane & 15);
  const int klo   = (lane & 16) ? 8 : 0;           // A-frag K offset (16-bit layout)
  const int koffB = (lane & 16) ? 8 : 0;           // B-frag K offset

  v8f acc[2][4] = {};

  for (int kb = 0; kb < KDIM; kb += 32) {
    { // stage A: 2 threads per row, 16 bf16 each (2x b128)
      int row = tid >> 1, half = tid & 1;
      const __bf16* s = wbf + (size_t)row * KDIM + kb + half * 16;
      __bf16* d = &As[row * 32 + half * 16];
      ((U4*)d)[0] = ((const U4*)s)[0];
      ((U4*)d)[1] = ((const U4*)s)[1];
    }
    // stage B transposed: vector global loads, scalar LDS scatter
#pragma unroll
    for (int it = 0; it < 2; ++it) {
      int task = tid + it * 256;                   // 512 tasks: 32 k x 16 segs
      int k = task >> 4, seg = task & 15;
      const __bf16* src = val + ((size_t)(kb + k)) * P1 + nBase + seg * 8;
      H8 h8;
      h8.q = *(const U4*)src;
#pragma unroll
      for (int i = 0; i < 8; ++i) VsT[(seg * 8 + i) * 32 + k] = h8.e[i];
      if (kb + 32 < KDIM)                          // prefetch next K-step rows
        __builtin_prefetch(src + (size_t)32 * P1, 0, 0);
    }
    __syncthreads();

    AFrag a0, a1;
    a0.q[0] = *(const U4*)&As[mRow0 * 32 + klo];
    a0.q[1] = *(const U4*)&As[mRow0 * 32 + 16 + klo];
    a1.q[0] = *(const U4*)&As[(mRow0 + 16) * 32 + klo];
    a1.q[1] = *(const U4*)&As[(mRow0 + 16) * 32 + 16 + klo];
#pragma unroll
    for (int ns = 0; ns < 4; ++ns) {
      AFrag bf;
      int n = (wn * 4 + ns) * 16 + (lane & 15);
      bf.q[0] = *(const U4*)&VsT[n * 32 + koffB];
      bf.q[1] = *(const U4*)&VsT[n * 32 + koffB + 16];
      acc[0][ns] = __builtin_amdgcn_wmma_f32_16x16x32_bf16(
          false, a0.v, false, bf.v, (short)0, acc[0][ns], false, false);
      acc[1][ns] = __builtin_amdgcn_wmma_f32_16x16x32_bf16(
          false, a1.v, false, bf.v, (short)0, acc[1][ns], false, false);
    }
    __syncthreads();
  }

  const int mHi = (lane & 16) ? 8 : 0;
#pragma unroll
  for (int mi = 0; mi < 2; ++mi) {
#pragma unroll
    for (int ns = 0; ns < 4; ++ns) {
      int col = nBase + (wn * 4 + ns) * 16 + (lane & 15);
#pragma unroll
      for (int r = 0; r < 8; ++r) {
        int o = wm * 32 + mi * 16 + mHi + r;
        y[((size_t)(b * COUT + o)) * P1 + col] = acc[mi][ns][r] + dcn_b[o];
      }
    }
  }
}

// ---------------------------------------------------------------------------
// 5) BatchNorm statistics: one block per channel, full reduction
// ---------------------------------------------------------------------------
__global__ __launch_bounds__(256)
void bn_stats_kernel(const float* __restrict__ src, float* __restrict__ stats,
                     int C, int nb, int shiftP) {
  __shared__ float s1[256], s2[256];
  const int c = blockIdx.x, tid = threadIdx.x;
  const long P = 1L << shiftP;
  const long total = (long)nb << shiftP;
  float sum = 0.f, sq = 0.f;
  for (long i = tid; i < total; i += 256) {
    long bi = i >> shiftP, p = i & (P - 1);
    float v = src[((bi * C + c) << shiftP) + p];
    sum += v; sq += v * v;
  }
  s1[tid] = sum; s2[tid] = sq; __syncthreads();
  for (int st = 128; st > 0; st >>= 1) {
    if (tid < st) { s1[tid] += s1[tid + st]; s2[tid] += s2[tid + st]; }
    __syncthreads();
  }
  if (tid == 0) {
    float m = s1[0] / (float)total;
    stats[c] = m;
    stats[C + c] = s2[0] / (float)total - m * m;
  }
}

// BN1 apply + ReLU -> bf16 deconv input
__global__ void bn1_apply_kernel(const float* __restrict__ y, const float* __restrict__ stats,
                                 const float* __restrict__ g, const float* __restrict__ bb,
                                 __bf16* __restrict__ ybf, int n) {
  int i = blockIdx.x * 256 + threadIdx.x;
  if (i >= n) return;
  int c = (i >> 12) & (COUT - 1);
  float m = stats[c], v = stats[COUT + c];
  float r = (y[i] - m) * rsqrtf(v + EPS) * g[c] + bb[c];
  ybf[i] = f2bf(fmaxf(r, 0.f));
}

// BN2 apply + ReLU in place on d_out
__global__ void bn2_apply_kernel(float* __restrict__ out, const float* __restrict__ stats,
                                 const float* __restrict__ g, const float* __restrict__ bb, int n) {
  int i = blockIdx.x * 256 + threadIdx.x;
  if (i >= n) return;
  int c = (i >> 14) & (COUT - 1);
  float m = stats[c], v = stats[COUT + c];
  float r = (out[i] - m) * rsqrtf(v + EPS) * g[c] + bb[c];
  out[i] = fmaxf(r, 0.f);
}

// ---------------------------------------------------------------------------
// 6) ConvTranspose2d(4x4, s2, p1) as 4 parity-class WMMA GEMMs.
//    Block = (b, cls, pair of output rows): N = 128 (2 rows x 64 cols),
//    wave = 2 M-tiles x 4 N-tiles. K = 512 (128 ch x 4 taps).
// ---------------------------------------------------------------------------
__global__ __launch_bounds__(256)
void deconv_gemm_kernel(const __bf16* __restrict__ w2, const __bf16* __restrict__ ybf,
                        float* __restrict__ out) {
  __shared__ __bf16 As[COUT * 32];
  __shared__ __bf16 VsT[128 * 32];

  const int bx = blockIdx.x;
  const int oyP = bx & 31;                 // pair of oy rows: oy = oyP*2 + (n>>6)
  const int cls = (bx >> 5) & 3;
  const int b   = bx >> 7;
  const int py = cls >> 1, px = cls & 1;

  const int tid  = threadIdx.x;
  const int lane = tid & 31;
  const int wv   = tid >> 5;
  const int wm   = wv & 3;
  const int wn   = wv >> 2;
  const int mRow0 = wm * 32 + (lane & 15);
  const int klo   = (lane & 16) ? 8 : 0;
  const int koffB = (lane & 16) ? 8 : 0;
  const __bf16* w2c = w2 + (size_t)cls * COUT * 512;

  v8f acc[2][4] = {};

  for (int kb = 0; kb < 512; kb += 32) {
    { // stage A
      int row = tid >> 1, half = tid & 1;
      const __bf16* s = w2c + (size_t)row * 512 + kb + half * 16;
      __bf16* d = &As[row * 32 + half * 16];
      ((U4*)d)[0] = ((const U4*)s)[0];
      ((U4*)d)[1] = ((const U4*)s)[1];
    }
    { // gather B tile: thread = (nLocal = tid>>1, K half = tid&1), 16 taps each
      int nLocal = tid >> 1, khalf = tid & 1;
      int oy = oyP * 2 + (nLocal >> 6);
      int ox = nLocal & 63;
      H8 lo, hi;
#pragma unroll
      for (int i = 0; i < 16; ++i) {
        int K = kb + khalf * 16 + i;
        int c = K >> 2, t = K & 3, tyi = t >> 1, txi = t & 1;
        int iy = oy + py - tyi;
        int ix = ox + px - txi;
        __bf16 v = f2bf(0.f);
        if (iy >= 0 && iy < HH && ix >= 0 && ix < WW)
          v = ybf[((size_t)(b * COUT + c)) * P1 + iy * WW + ix];
        if (i < 8) lo.e[i] = v; else hi.e[i - 8] = v;
      }
      *(U4*)&VsT[nLocal * 32 + khalf * 16]     = lo.q;
      *(U4*)&VsT[nLocal * 32 + khalf * 16 + 8] = hi.q;
    }
    __syncthreads();

    AFrag a0, a1;
    a0.q[0] = *(const U4*)&As[mRow0 * 32 + klo];
    a0.q[1] = *(const U4*)&As[mRow0 * 32 + 16 + klo];
    a1.q[0] = *(const U4*)&As[(mRow0 + 16) * 32 + klo];
    a1.q[1] = *(const U4*)&As[(mRow0 + 16) * 32 + 16 + klo];
#pragma unroll
    for (int ns = 0; ns < 4; ++ns) {
      AFrag bf;
      int n = (wn * 4 + ns) * 16 + (lane & 15);
      bf.q[0] = *(const U4*)&VsT[n * 32 + koffB];
      bf.q[1] = *(const U4*)&VsT[n * 32 + koffB + 16];
      acc[0][ns] = __builtin_amdgcn_wmma_f32_16x16x32_bf16(
          false, a0.v, false, bf.v, (short)0, acc[0][ns], false, false);
      acc[1][ns] = __builtin_amdgcn_wmma_f32_16x16x32_bf16(
          false, a1.v, false, bf.v, (short)0, acc[1][ns], false, false);
    }
    __syncthreads();
  }

  const int mHi = (lane & 16) ? 8 : 0;
#pragma unroll
  for (int mi = 0; mi < 2; ++mi) {
#pragma unroll
    for (int ns = 0; ns < 4; ++ns) {
      int n = (wn * 4 + ns) * 16 + (lane & 15);
      int oy = oyP * 2 + (n >> 6);
      int ox = n & 63;
      int Y = 2 * oy + py;
      int X = 2 * ox + px;
#pragma unroll
      for (int r = 0; r < 8; ++r) {
        int o = wm * 32 + mi * 16 + mHi + r;
        out[((size_t)(b * COUT + o)) * P2 + Y * WO + X] = acc[mi][ns][r];
      }
    }
  }
}

// ---------------------------------------------------------------------------
// Host launcher
// ---------------------------------------------------------------------------
extern "C" void kernel_launch(void* const* d_in, const int* in_sizes, int n_in,
                              void* d_out, int out_size, void* d_ws, size_t ws_size,
                              hipStream_t stream) {
  const float* x     = (const float*)d_in[0];
  const float* off_w = (const float*)d_in[1];
  const float* off_b = (const float*)d_in[2];
  const float* dcn_w = (const float*)d_in[3];
  const float* dcn_b = (const float*)d_in[4];
  const float* bn1_g = (const float*)d_in[5];
  const float* bn1_b = (const float*)d_in[6];
  const float* up_w  = (const float*)d_in[7];
  const float* bn2_g = (const float*)d_in[8];
  const float* bn2_b = (const float*)d_in[9];
  float* out = (float*)d_out;

  // workspace layout (all 256-aligned), total ~49 MB
  char* ws = (char*)d_ws;
  const size_t OFF_OM  = 0;                                    // 8*27*4096*4
  const size_t OFF_VAL = OFF_OM  + (size_t)NB * 27 * P1 * 4;   // 2304*4096*2
  const size_t OFF_WBF = OFF_VAL + (size_t)KDIM * P1 * 2;      // 128*2304*2
  const size_t OFF_W2  = OFF_WBF + (size_t)COUT * KDIM * 2;    // 4*128*512*2
  const size_t OFF_Y   = OFF_W2  + (size_t)4 * COUT * 512 * 2; // 8*128*4096*4
  const size_t OFF_YBF = OFF_Y   + (size_t)NB * COUT * P1 * 4; // 8*128*4096*2
  const size_t OFF_S1  = OFF_YBF + (size_t)NB * COUT * P1 * 2; // 1024
  const size_t OFF_S2  = OFF_S1  + 1024;

  float*  om     = (float*)(ws + OFF_OM);
  __bf16* val    = (__bf16*)(ws + OFF_VAL);
  __bf16* wbf    = (__bf16*)(ws + OFF_WBF);
  __bf16* w2     = (__bf16*)(ws + OFF_W2);
  float*  ybuf   = (float*)(ws + OFF_Y);
  __bf16* ybf    = (__bf16*)(ws + OFF_YBF);
  float*  stats1 = (float*)(ws + OFF_S1);
  float*  stats2 = (float*)(ws + OFF_S2);

  // 1) offset conv
  offconv_kernel<<<dim3(16, NB), 256, 0, stream>>>(x, off_w, off_b, om);

  // 2) weight conversions
  convert_w_kernel<<<(COUT * KDIM + 255) / 256, 256, 0, stream>>>(dcn_w, wbf, COUT * KDIM);
  build_w2_kernel<<<(4 * COUT * 512 + 255) / 256, 256, 0, stream>>>(up_w, w2);

  // 3+4) per-batch deformable im2col + WMMA GEMM (val buffer reused)
  for (int b = 0; b < NB; ++b) {
    build_val_kernel<<<(9 * P1 + 255) / 256, 256, 0, stream>>>(x, om, val, b);
    gemm_dcn_kernel<<<P1 / 128, 256, 0, stream>>>(wbf, val, dcn_b, ybuf, b);
  }

  // 5) BN1 + ReLU -> bf16
  bn_stats_kernel<<<COUT, 256, 0, stream>>>(ybuf, stats1, COUT, NB, 12);
  bn1_apply_kernel<<<(NB * COUT * P1 + 255) / 256, 256, 0, stream>>>(
      ybuf, stats1, bn1_g, bn1_b, ybf, NB * COUT * P1);

  // 6) deconv via WMMA (b x 4 classes x 32 row-pairs)
  deconv_gemm_kernel<<<NB * 4 * 32, 256, 0, stream>>>(w2, ybf, out);

  // 7) BN2 + ReLU in place
  bn_stats_kernel<<<COUT, 256, 0, stream>>>(out, stats2, COUT, NB, 14);
  bn2_apply_kernel<<<(NB * COUT * P2 + 255) / 256, 256, 0, stream>>>(
      out, stats2, bn2_g, bn2_b, NB * COUT * P2);
}
